// MultiHeadAttention_42494406427456
// MI455X (gfx1250) — matrix operand order repaired and checked by reading, not compiled, not run on hardware
//
#include <hip/hip_runtime.h>

// ---------------------------------------------------------------------------
// MI455X (gfx1250, wave32) multi-head cross-attention with RoPE.
// bf16 WMMA (v_wmma_f32_16x16x32_bf16) for all GEMMs, f32 accumulation.
// W-tile staging in the projection GEMMs uses the Tensor Data Mover
// (tensor_load_to_lds + s_wait_tensorcnt) when the builtin is available,
// software-pipelined so the DMA overlaps the WMMA work of the current tile.
// ---------------------------------------------------------------------------

typedef __attribute__((ext_vector_type(16))) __bf16 v16bf;
typedef __attribute__((ext_vector_type(8)))  __bf16 v8bf;
typedef __attribute__((ext_vector_type(8)))  float  v8f;
typedef __attribute__((ext_vector_type(4)))  float  v4f;

typedef unsigned int u32x4 __attribute__((ext_vector_type(4)));
typedef int          i32x4 __attribute__((ext_vector_type(4)));
typedef int          i32x8 __attribute__((ext_vector_type(8)));

#define C_B   2
#define C_SEQ 2048      // Q == KV == 2048
#define C_H   1024
#define C_NH  16
#define C_HD  64

#if defined(__has_builtin)
#if __has_builtin(__builtin_amdgcn_tensor_load_to_lds) && \
    __has_builtin(__builtin_amdgcn_s_wait_tensorcnt)
#define HAVE_TDM 1
#endif
#endif
#ifndef HAVE_TDM
#define HAVE_TDM 0
#endif

__device__ __forceinline__ v8f wmma_bf16(v16bf a, v16bf b, v8f c) {
  // D = A(16x32) * B(32x16) + C(16x16 f32)
  return __builtin_amdgcn_wmma_f32_16x16x32_bf16(
      /*neg_a=*/false, a, /*neg_b=*/false, b,
      /*c_mod=*/(short)0, c, /*reuse_a=*/false, /*reuse_b=*/false);
}

__device__ __forceinline__ v16bf pack16(v8bf lo, v8bf hi) {
  v16bf r;
#pragma unroll
  for (int i = 0; i < 8; ++i) { r[i] = lo[i]; r[8 + i] = hi[i]; }
  return r;
}

__device__ __forceinline__ v16bf cvt16(v4f a, v4f b, v4f c, v4f d) {
  v16bf r;
#pragma unroll
  for (int i = 0; i < 4; ++i) {
    r[i]      = (__bf16)a[i];
    r[4 + i]  = (__bf16)b[i];
    r[8 + i]  = (__bf16)c[i];
    r[12 + i] = (__bf16)d[i];
  }
  return r;
}

#if HAVE_TDM
// ---------------------------------------------------------------------------
// TDM: DMA a [rows x cols] f32 tile (row stride `stride_elems`) from global
// memory into LDS at byte offset lds_off. Descriptor per CDNA5 ISA ch.8:
//   group0: count=1 | lds_addr | global_addr[56:0] | type=2
//   group1: data_size=4B, tensor_dim0/1, tile_dim0/1, tensor_dim0_stride
//   group2/3: zero (<=2D tensor, no iterate/gather)
// All inputs are wave-uniform -> groups legalize into SGPRs.
// ---------------------------------------------------------------------------
__device__ __forceinline__ void tdm_load_tile_f32(
    const float* gsrc, unsigned lds_off, int rows, int cols, int stride_elems)
{
  const unsigned long long ga = (unsigned long long)(size_t)(const void*)gsrc;
  u32x4 g0;
  g0[0] = 1u;                                                 // count=1 (valid)
  g0[1] = lds_off;                                            // lds_addr
  g0[2] = (unsigned)ga;                                       // global_addr lo
  g0[3] = (unsigned)((ga >> 32) & 0x1FFFFFFu) | (2u << 30);   // addr hi | type=2

  const unsigned td0 = (unsigned)stride_elems;                // tensor width
  i32x8 g1;
  g1[0] = 2 << 16;                                            // data_size=4B
  g1[1] = (int)((td0 & 0xFFFFu) << 16);                       // tensor_dim0 lo
  g1[2] = (int)(((td0 >> 16) & 0xFFFFu) | (0x8000u << 16));   // dim0 hi|dim1 lo
  g1[3] = (int)(((unsigned)cols & 0xFFFFu) << 16);            // dim1 hi|tile_dim0
  g1[4] = rows & 0xFFFF;                                      // tile_dim1 (dim2=0)
  g1[5] = (int)td0;                                           // dim0_stride lo
  g1[6] = 0;                                                  // stride hi / dim1_stride lo
  g1[7] = 0;

  const i32x4 z4 = {0, 0, 0, 0};
#if __clang_major__ >= 23
  const i32x8 z8 = {0, 0, 0, 0, 0, 0, 0, 0};
  __builtin_amdgcn_tensor_load_to_lds(g0, g1, z4, z4, z8, 0);
#else
  __builtin_amdgcn_tensor_load_to_lds(g0, g1, z4, z4, 0);
#endif
}
#endif  // HAVE_TDM

// ---------------------------------------------------------------------------
// Generic 128x128-tile bf16-WMMA GEMM: C = A[M,1024] * W[1024,1024] + bias
// 8 waves/block; wave = 32(M) x 64(N) = 2x4 grid of 16x16 WMMA tiles.
// A fragments load straight from global f32 (converted in-register);
// W tiles arrive in LDS via TDM (f32), are transposed+converted to bf16 in
// LDS so B fragments are two k-contiguous 16B reads per lane.
// MODE: 0=Q-proj(+RoPE), 1=K-proj(+RoPE), 2=V-proj(writes V^T), 3=out-proj.
// ---------------------------------------------------------------------------
#define PROJ_Q 0
#define PROJ_K 1
#define PROJ_V 2
#define PROJ_O 3

#define WT_STRIDE 40   // bf16 elems per LDS row (32 + 8 pad), keeps 16B align

template <int MODE>
__global__ __launch_bounds__(256) void proj_kernel(
    const float* __restrict__ A,       // [4096, 1024] activations
    const float* __restrict__ W,       // [1024, 1024]
    const float* __restrict__ bias,    // [1024]
    const float* __restrict__ freqs,   // [2048, 64] (Q/K modes)
    const float* __restrict__ resid,   // [4096, 1024] (O mode)
    float*  __restrict__ out_f32,      // heads f32 / embeddings
    __bf16* __restrict__ out_bf16)     // qrot / krot / vT workspace
{
  __shared__ __bf16 ldsWT[128 * WT_STRIDE];
#if HAVE_TDM
  __shared__ float  ldsWf32[32 * 128];   // TDM staging tile (16 KB)
#endif

  const int tid  = threadIdx.x;
  const int w    = tid >> 5;
  const int lane = tid & 31;
  const int l15  = lane & 15;
  const int lh   = lane >> 4;

  const int n_blk = blockIdx.x * 128;
  const int m_blk = blockIdx.y * 128;
  const int mw    = m_blk + (w >> 1) * 32;   // 2 M-tiles
  const int nw    = n_blk + (w & 1) * 64;    // 4 N-tiles (one full head width)

  v8f acc[2][4] = {};

#if HAVE_TDM
  const unsigned ldsW_off = (unsigned)(size_t)(void*)ldsWf32;
  if (w == 0)  // uniform branch: one wave per block drives the TDM
    tdm_load_tile_f32(W + n_blk, ldsW_off, 32, 128, C_H);
#endif

  for (int k0 = 0; k0 < C_H; k0 += 32) {
#if HAVE_TDM
    if (w == 0) __builtin_amdgcn_s_wait_tensorcnt(0);
    __syncthreads();                      // tile visible to all waves
    float stage[16];                      // drain f32 tile into registers
#pragma unroll
    for (int rep = 0; rep < 16; ++rep)
      stage[rep] = ldsWf32[rep * 256 + tid];
    __syncthreads();                      // staging buffer free again
    if (w == 0 && k0 + 32 < C_H)          // pipeline: DMA next tile now
      tdm_load_tile_f32(W + (size_t)(k0 + 32) * C_H + n_blk, ldsW_off, 32, 128, C_H);
#pragma unroll
    for (int rep = 0; rep < 16; ++rep) {  // regs -> transposed bf16 LDS
      const int e = rep * 256 + tid;
      ldsWT[(e & 127) * WT_STRIDE + (e >> 7)] = (__bf16)stage[rep];
    }
    __syncthreads();
#else
    __syncthreads();
#pragma unroll
    for (int rep = 0; rep < 16; ++rep) {
      const int e  = rep * 256 + tid;
      const int kk = e >> 7;
      const int nn = e & 127;
      ldsWT[nn * WT_STRIDE + kk] = (__bf16)W[(size_t)(k0 + kk) * C_H + n_blk + nn];
    }
    __syncthreads();
#endif

    // A fragments (direct from global f32, per ISA 16x32 bf16 layout).
    v16bf afr[2];
#pragma unroll
    for (int i = 0; i < 2; ++i) {
      const float* ap = A + (size_t)(mw + i * 16 + l15) * C_H + k0 + lh * 8;
      v4f x0 = *(const v4f*)(ap);
      v4f x1 = *(const v4f*)(ap + 4);
      v4f x2 = *(const v4f*)(ap + 16);
      v4f x3 = *(const v4f*)(ap + 20);
      afr[i] = cvt16(x0, x1, x2, x3);
    }

    // B fragments from LDS + WMMA chain.
#pragma unroll
    for (int j = 0; j < 4; ++j) {
      const __bf16* bp = &ldsWT[((w & 1) * 64 + j * 16 + l15) * WT_STRIDE + lh * 8];
      v16bf bfr = pack16(*(const v8bf*)bp, *(const v8bf*)(bp + 16));
#pragma unroll
      for (int i = 0; i < 2; ++i)
        acc[i][j] = wmma_bf16(afr[i], bfr, acc[i][j]);
    }
  }

  // ---- epilogue: bias -------------------------------------------------
#pragma unroll
  for (int j = 0; j < 4; ++j) {
    const float bn = bias[nw + j * 16 + l15];
#pragma unroll
    for (int i = 0; i < 2; ++i)
#pragma unroll
      for (int r = 0; r < 8; ++r)
        acc[i][j][r] += bn;
  }

  if constexpr (MODE == PROJ_Q || MODE == PROJ_K) {
    // write pre-RoPE heads f32, and RoPE'd bf16 into workspace.
#pragma unroll
    for (int i = 0; i < 2; ++i) {
#pragma unroll
      for (int r = 0; r < 8; ++r) {
        const int m   = mw + i * 16 + r + 8 * lh;
        const int b   = m >> 11;
        const int pos = m & (C_SEQ - 1);
#pragma unroll
        for (int j = 0; j < 4; ++j) {
          const int n = nw + j * 16 + l15;
          const int h = n >> 6;
          const int d = n & 63;
          const float c = acc[i][j][r];
          const size_t base = ((size_t)(b * C_NH + h) * C_SEQ + pos) * C_HD + d;
          out_f32[base] = c;
          float sf, cf;
          __sincosf(freqs[pos * C_HD + d], &sf, &cf);
          const float part = acc[i][(j + 2) & 3][r];
          const float rot  = (j < 2) ? -part : part;   // rotate_half
          out_bf16[base] = (__bf16)(c * cf + rot * sf);
        }
      }
    }
  } else if constexpr (MODE == PROJ_V) {
#pragma unroll
    for (int i = 0; i < 2; ++i) {
#pragma unroll
      for (int r = 0; r < 8; ++r) {
        const int m   = mw + i * 16 + r + 8 * lh;
        const int b   = m >> 11;
        const int pos = m & (C_SEQ - 1);
#pragma unroll
        for (int j = 0; j < 4; ++j) {
          const int n = nw + j * 16 + l15;
          const int h = n >> 6;
          const int d = n & 63;
          const float c = acc[i][j][r];
          out_f32[((size_t)(b * C_NH + h) * C_SEQ + pos) * C_HD + d] = c;
          // V^T [b,h,d,kv] so PV B-fragments are contiguous in kv.
          out_bf16[((size_t)(b * C_NH + h) * C_HD + d) * C_SEQ + pos] = (__bf16)c;
        }
      }
    }
  } else {  // PROJ_O: embeddings = ctx@Wd + bd + hidden
#pragma unroll
    for (int i = 0; i < 2; ++i) {
#pragma unroll
      for (int r = 0; r < 8; ++r) {
        const size_t m = mw + i * 16 + r + 8 * lh;
#pragma unroll
        for (int j = 0; j < 4; ++j) {
          const int n = nw + j * 16 + l15;
          out_f32[m * C_H + n] = acc[i][j][r] + resid[m * C_H + n];
        }
      }
    }
  }
}

// ---------------------------------------------------------------------------
// Attention: per block = one (b,h) pair x 128 query rows, 8 waves.
// Each wave owns 16 complete rows (full KV) -> no cross-wave reductions.
// Pass 1: scores via WMMA, write scores, online row max/sum.
// Pass 2: recompute scores from L2-resident bf16 Q/K (cheaper than
//         re-reading 536MB of scores from HBM), write probs, PV via WMMA
//         with a per-wave LDS C->A fragment relayout.
// ---------------------------------------------------------------------------
#define P_STRIDE 136   // bf16 elems per ldsP row (128 + 8 pad)

__global__ __launch_bounds__(256) void attn_kernel(
    const __bf16* __restrict__ qrot,   // [b,h,q,d]
    const __bf16* __restrict__ krot,   // [b,h,kv,d]
    const __bf16* __restrict__ vT,     // [b,h,d,kv]
    const float*  __restrict__ amask,  // [b,kv]
    float* __restrict__ scores_out,    // [b,h,q,kv]
    float* __restrict__ probs_out,     // [b,h,q,kv]
    float* __restrict__ ctx_out)       // [b,q, h*64+d]
{
  __shared__ float  ldsBias[C_SEQ];
  __shared__ __bf16 ldsP[8 * 16 * P_STRIDE];

  const int tid  = threadIdx.x;
  const int w    = tid >> 5;
  const int lane = tid & 31;
  const int l15  = lane & 15;
  const int lh   = lane >> 4;

  const int bh = blockIdx.x;
  const int b  = bh >> 4;
  const int h  = bh & 15;
  const int q0 = blockIdx.y * 128 + w * 16;   // wave's 16 rows

  // mask -> additive bias, staged once in LDS
  for (int i = tid; i < C_SEQ; i += 256)
    ldsBias[i] = (amask[b * C_SEQ + i] > 0.5f) ? 0.0f : -1.0e30f;
  __syncthreads();

  // Q fragments for this wave's 16 rows (d = 0..63 -> two K=32 fragments),
  // held in registers for the whole kernel.
  v16bf aq[2];
#pragma unroll
  for (int ki = 0; ki < 2; ++ki) {
    const __bf16* qp = qrot + ((size_t)bh * C_SEQ + q0 + l15) * C_HD + ki * 32 + lh * 16;
    aq[ki] = pack16(*(const v8bf*)qp, *(const v8bf*)(qp + 8));
  }

  float mrow[8], lrow[8];
#pragma unroll
  for (int r = 0; r < 8; ++r) { mrow[r] = -3.0e38f; lrow[r] = 0.0f; }

  // ---------------- pass 1: scores + online softmax stats ----------------
  for (int kv0 = 0; kv0 < C_SEQ; kv0 += 128) {
    v8f s[8] = {};
#pragma unroll
    for (int j = 0; j < 8; ++j) {
#pragma unroll
      for (int ki = 0; ki < 2; ++ki) {
        const __bf16* kp =
            krot + ((size_t)bh * C_SEQ + kv0 + j * 16 + l15) * C_HD + ki * 32 + lh * 16;
        __builtin_prefetch(kp + 128 * C_HD, 0, 1);
        v16bf bfr = pack16(*(const v8bf*)kp, *(const v8bf*)(kp + 8));
        s[j] = wmma_bf16(aq[ki], bfr, s[j]);
      }
    }
#pragma unroll
    for (int r = 0; r < 8; ++r) {
      const int row = q0 + r + 8 * lh;
      float vals[8];
      float rowm = -3.0e38f;
#pragma unroll
      for (int j = 0; j < 8; ++j) {
        const int kv = kv0 + j * 16 + l15;
        const float v = s[j][r] * 0.125f + ldsBias[kv];
        vals[j] = v;
        scores_out[((size_t)bh * C_SEQ + row) * C_SEQ + kv] = v;
        rowm = fmaxf(rowm, v);
      }
#pragma unroll
      for (int off = 1; off < 16; off <<= 1)
        rowm = fmaxf(rowm, __shfl_xor(rowm, off, 32));
      const float mnew = fmaxf(mrow[r], rowm);
      float esum = 0.0f;
#pragma unroll
      for (int j = 0; j < 8; ++j) esum += __expf(vals[j] - mnew);
#pragma unroll
      for (int off = 1; off < 16; off <<= 1)
        esum += __shfl_xor(esum, off, 32);
      lrow[r] = lrow[r] * __expf(mrow[r] - mnew) + esum;
      mrow[r] = mnew;
    }
  }

  float rl[8];
#pragma unroll
  for (int r = 0; r < 8; ++r) rl[r] = 1.0f / lrow[r];

  // ---------------- pass 2: probs + PV -----------------------------------
  v8f ctx[4] = {};
  __bf16* myP = ldsP + w * 16 * P_STRIDE;   // wave-private; DS ops in-order

  for (int kv0 = 0; kv0 < C_SEQ; kv0 += 128) {
    v8f s[8] = {};
#pragma unroll
    for (int j = 0; j < 8; ++j) {
#pragma unroll
      for (int ki = 0; ki < 2; ++ki) {
        const __bf16* kp =
            krot + ((size_t)bh * C_SEQ + kv0 + j * 16 + l15) * C_HD + ki * 32 + lh * 16;
        v16bf bfr = pack16(*(const v8bf*)kp, *(const v8bf*)(kp + 8));
        s[j] = wmma_bf16(aq[ki], bfr, s[j]);
      }
    }
    // normalized probs: write f32 out + bf16 into LDS (C-layout -> A-layout)
#pragma unroll
    for (int j = 0; j < 8; ++j) {
      const float bj = ldsBias[kv0 + j * 16 + l15];
#pragma unroll
      for (int r = 0; r < 8; ++r) {
        const int row = q0 + r + 8 * lh;
        const float p = __expf(s[j][r] * 0.125f + bj - mrow[r]) * rl[r];
        probs_out[((size_t)bh * C_SEQ + row) * C_SEQ + kv0 + j * 16 + l15] = p;
        myP[(r + 8 * lh) * P_STRIDE + j * 16 + l15] = (__bf16)p;
      }
    }
    // P(16x128) @ V(128x64): 4 K-chunks x 4 N-tiles of WMMA
#pragma unroll
    for (int kk = 0; kk < 4; ++kk) {
      const __bf16* pp = myP + l15 * P_STRIDE + kk * 32 + lh * 16;
      v16bf pf = pack16(*(const v8bf*)pp, *(const v8bf*)(pp + 8));
#pragma unroll
      for (int nt = 0; nt < 4; ++nt) {
        const __bf16* vp =
            vT + ((size_t)bh * C_HD + nt * 16 + l15) * C_SEQ + kv0 + kk * 32 + lh * 16;
        v16bf vf = pack16(*(const v8bf*)vp, *(const v8bf*)(vp + 8));
        ctx[nt] = wmma_bf16(pf, vf, ctx[nt]);
      }
    }
  }

  // ctx -> [b, q, h*64 + d] (f32), consumed by out-projection GEMM
#pragma unroll
  for (int nt = 0; nt < 4; ++nt) {
    const int hcol = h * C_HD + nt * 16 + l15;
#pragma unroll
    for (int r = 0; r < 8; ++r) {
      const int row = q0 + r + 8 * lh;
      ctx_out[((size_t)b * C_SEQ + row) * C_H + hcol] = ctx[nt][r];
    }
  }
}

// ---------------------------------------------------------------------------
extern "C" void kernel_launch(void* const* d_in, const int* in_sizes, int n_in,
                              void* d_out, int out_size, void* d_ws, size_t ws_size,
                              hipStream_t stream) {
  const float* hidden  = (const float*)d_in[0];
  const float* encoder = (const float*)d_in[1];
  const float* amask   = (const float*)d_in[2];
  const float* qf      = (const float*)d_in[3];
  const float* kf      = (const float*)d_in[4];
  const float* Wq      = (const float*)d_in[5];
  const float* bq      = (const float*)d_in[6];
  const float* Wk      = (const float*)d_in[7];
  const float* bk      = (const float*)d_in[8];
  const float* Wv      = (const float*)d_in[9];
  const float* bv      = (const float*)d_in[10];
  const float* Wd      = (const float*)d_in[11];
  const float* bd      = (const float*)d_in[12];

  const size_t SZ_H  = (size_t)C_B * C_NH * C_SEQ * C_HD;   // 4,194,304
  const size_t SZ_QK = (size_t)C_B * C_NH * C_SEQ * C_SEQ;  // 134,217,728

  // tuple order: (embeddings, q, v, k, probs, scores, ctx)
  float* out        = (float*)d_out;
  float* out_emb    = out;
  float* out_q      = out_emb + SZ_H;
  float* out_v      = out_q + SZ_H;
  float* out_k      = out_v + SZ_H;
  float* out_probs  = out_k + SZ_H;
  float* out_scores = out_probs + SZ_QK;
  float* out_ctx    = out_scores + SZ_QK;

  __bf16* ws_qrot = (__bf16*)d_ws;        // 8 MB
  __bf16* ws_krot = ws_qrot + SZ_H;       // 8 MB
  __bf16* ws_vT   = ws_krot + SZ_H;       // 8 MB

  const dim3 blk(256);
  const dim3 gproj(C_H / 128, (C_B * C_SEQ) / 128);  // (8, 32)

  proj_kernel<PROJ_Q><<<gproj, blk, 0, stream>>>(hidden,  Wq, bq, qf, nullptr, out_q, ws_qrot);
  proj_kernel<PROJ_K><<<gproj, blk, 0, stream>>>(encoder, Wk, bk, kf, nullptr, out_k, ws_krot);
  proj_kernel<PROJ_V><<<gproj, blk, 0, stream>>>(encoder, Wv, bv, nullptr, nullptr, out_v, ws_vT);

  attn_kernel<<<dim3(C_B * C_NH, C_SEQ / 128), blk, 0, stream>>>(
      ws_qrot, ws_krot, ws_vT, amask, out_scores, out_probs, out_ctx);

  proj_kernel<PROJ_O><<<gproj, blk, 0, stream>>>(out_ctx, Wd, bd, nullptr, hidden, out_emb, nullptr);
}
